// ScaledDotProductAttention_44796508897312
// MI455X (gfx1250) — matrix operand order repaired and checked by reading, not compiled, not run on hardware
//
#include <hip/hip_runtime.h>

// ---------------------------------------------------------------------------
// Types for CDNA5 WMMA (wave32): A/B = 16 bf16 (8 VGPRs), C/D = 8 f32.
// ---------------------------------------------------------------------------
typedef __bf16 bf16_t;
typedef __attribute__((ext_vector_type(16))) __bf16 v16bf;
typedef __attribute__((ext_vector_type(8)))  __bf16 v8bf;
typedef __attribute__((ext_vector_type(8)))  float  v8f;

__device__ __forceinline__ v8bf ld8(const bf16_t* p) {
    return *(const v8bf*)p;                      // 16B global/LDS load (b128)
}
__device__ __forceinline__ v16bf cat(v8bf lo, v8bf hi) {
    return __builtin_shufflevector(lo, hi, 0,1,2,3,4,5,6,7,8,9,10,11,12,13,14,15);
}
__device__ __forceinline__ v8f wmma_bf16(v16bf a, v16bf b, v8f c) {
    // D = A(16x32 bf16) x B(32x16 bf16) + C(16x16 f32)
    return __builtin_amdgcn_wmma_f32_16x16x32_bf16(false, a, false, b,
                                                   (short)0, c, false, false);
}

// Problem constants (B=2, H=16, S=2048, D=64)
#define NB  2
#define NH  16
#define NS  2048
#define ND  64

// log2(e)/TEMPERATURE : scores come out of WMMA already in the log2 domain,
// so every softmax exp is a single v_exp_f32 (exp2), no log2e multiplies.
#define QSCALE (0.125f * 1.44269504088896340736f)

// ---------------------------------------------------------------------------
// Prep kernel: q -> bf16 scaled by log2e/8, k -> bf16 row-major,
// v -> bf16 TRANSPOSED to [B*H, D, S] so the P@V WMMA B-operand is K-contiguous.
// Weights staged as log2(mask ? sent_att : 0):  masked -> -inf, exp2 -> exact 0,
// and the sent_att reweight becomes an ADD inside the exponent.
// ---------------------------------------------------------------------------
__global__ void prep_kernel(const float* __restrict__ q, const float* __restrict__ k,
                            const float* __restrict__ v, const float* __restrict__ sent,
                            const int* __restrict__ mask,
                            bf16_t* __restrict__ qb, bf16_t* __restrict__ kb,
                            bf16_t* __restrict__ vt, float* __restrict__ wm) {
    const size_t i = (size_t)blockIdx.x * blockDim.x + threadIdx.x;
    constexpr size_t N = (size_t)NB * NH * NS * ND;
    if (i >= N) return;
    qb[i] = (bf16_t)(q[i] * QSCALE);
    kb[i] = (bf16_t)k[i];
    const size_t s  = i & (NS - 1);              // output linear idx = [bh][d][s]
    const size_t d  = (i >> 11) & (ND - 1);
    const size_t bh = i >> 17;
    vt[i] = (bf16_t)v[(bh * NS + s) * ND + d];
    if (i < (size_t)NB * NS)
        wm[i] = mask[i] ? __builtin_log2f(sent[i]) : -__builtin_inff();
}

// ---------------------------------------------------------------------------
// Fused attention: 1 WG = 64 query rows of one (b,h); 4 waves x 16 rows.
//   Pass 1a: QK^T, per-cell running MAX only (no trans ops).
//   Pass 1b: QK^T again, rowsum l += exp2(c + u - m); plain add-reduce.
//   Pass 2 : QK^T again, attn = exp2(c + u + t2) with t2 = -m - log2(l);
//            non-temporal store, P->LDS transpose, P@V WMMA accumulate.
// Score recompute is deliberate: WMMA is far from its roofline while the
// TRANS pipe and the 536MB attn write are the binding limits.
// ---------------------------------------------------------------------------
__global__ __launch_bounds__(128)
void attn_fused(const bf16_t* __restrict__ qb, const bf16_t* __restrict__ kb,
                const bf16_t* __restrict__ vt, const float* __restrict__ wm,
                float* __restrict__ outO, float* __restrict__ outA) {
    __shared__ float  llw[NS];                   // log2(mask*sent) for this batch
    __shared__ bf16_t lp[4][16][32];             // per-wave P tile (row-major)

    const int wid  = threadIdx.x >> 5;
    const int lane = threadIdx.x & 31;
    const int lo16 = lane & 15;
    const int hi   = lane >> 4;                  // 0: lanes 0-15, 1: lanes 16-31

    const int nQT = NS / 64;
    const int bh  = blockIdx.x / nQT;
    const int qt  = blockIdx.x % nQT;
    const int b   = bh >> 4;
    const int q0  = qt * 64 + wid * 16;

    {   // stage log-weights into LDS (8 KB)
        const float4* src = (const float4*)(wm + (size_t)b * NS);
        for (int i = threadIdx.x; i < NS / 4; i += 128)
            ((float4*)llw)[i] = src[i];
    }
    __syncthreads();

    // --- Q A-fragments, resident for the whole kernel -----------------------
    // A 16x32 in this code's contiguous k order (k = 16*hi + s): lane L holds row L%16; hi-half holds K = 16..31.
    const bf16_t* qrow = qb + ((size_t)bh * NS + q0 + lo16) * ND;
    const v16bf a0 = cat(ld8(qrow +      hi * 16), ld8(qrow +      hi * 16 + 8)); // d 0..31
    const v16bf a1 = cat(ld8(qrow + 32 + hi * 16), ld8(qrow + 32 + hi * 16 + 8)); // d 32..63

    const bf16_t* kbase = kb + (size_t)bh * NS * ND;

    // ---------------- Pass 1a: row max (log2-domain scores) -----------------
    float m[8];
#pragma unroll
    for (int i = 0; i < 8; ++i) m[i] = -3.0e38f;

    for (int kt = 0; kt < NS / 16; ++kt) {
        const bf16_t* krow = kbase + (size_t)(kt * 16 + lo16) * ND;
        __builtin_prefetch(krow + 16 * ND, 0, 1);          // global_prefetch_b8
        v16bf b0 = cat(ld8(krow +      hi * 16), ld8(krow +      hi * 16 + 8));
        v16bf b1 = cat(ld8(krow + 32 + hi * 16), ld8(krow + 32 + hi * 16 + 8));
        v8f c = {};
        c = wmma_bf16(a0, b0, c);
        c = wmma_bf16(a1, b1, c);
#pragma unroll
        for (int i = 0; i < 8; ++i) m[i] = fmaxf(m[i], c[i]);
    }
#pragma unroll
    for (int i = 0; i < 8; ++i) {
#pragma unroll
        for (int st = 1; st < 16; st <<= 1)
            m[i] = fmaxf(m[i], __shfl_xor(m[i], st, 16));
    }

    // ---------------- Pass 1b: weighted row sum ------------------------------
    float l[8];
#pragma unroll
    for (int i = 0; i < 8; ++i) l[i] = 0.0f;

    for (int kt = 0; kt < NS / 16; ++kt) {
        const bf16_t* krow = kbase + (size_t)(kt * 16 + lo16) * ND;
        v16bf b0 = cat(ld8(krow +      hi * 16), ld8(krow +      hi * 16 + 8));
        v16bf b1 = cat(ld8(krow + 32 + hi * 16), ld8(krow + 32 + hi * 16 + 8));
        v8f c = {};
        c = wmma_bf16(a0, b0, c);
        c = wmma_bf16(a1, b1, c);
        const float u = llw[kt * 16 + lo16];     // per-lane: one key per lane
#pragma unroll
        for (int i = 0; i < 8; ++i)
            l[i] += __builtin_exp2f((c[i] + u) - m[i]);   // bare v_exp_f32
    }
    float t2[8];
#pragma unroll
    for (int i = 0; i < 8; ++i) {
#pragma unroll
        for (int st = 1; st < 16; st <<= 1)
            l[i] += __shfl_xor(l[i], st, 16);    // plain add-reduce, no trans
        t2[i] = -m[i] - __builtin_log2f(l[i]);   // p = 2^(c + u + t2)
    }

    // ---------------- Pass 2: emit attn + accumulate O -----------------------
    float* aOut = outA + ((size_t)bh * NS + q0) * NS;
    v8f o0 = {}, o1 = {}, o2 = {}, o3 = {};

    for (int kt2 = 0; kt2 < NS / 32; ++kt2) {
        const int k0p = kt2 * 32;
#pragma unroll
        for (int sub = 0; sub < 2; ++sub) {
            const int k0 = k0p + sub * 16;
            const bf16_t* krow = kbase + (size_t)(k0 + lo16) * ND;
            v16bf b0 = cat(ld8(krow +      hi * 16), ld8(krow +      hi * 16 + 8));
            v16bf b1 = cat(ld8(krow + 32 + hi * 16), ld8(krow + 32 + hi * 16 + 8));
            v8f c = {};
            c = wmma_bf16(a0, b0, c);
            c = wmma_bf16(a1, b1, c);
            const float u = llw[k0 + lo16];
#pragma unroll
            for (int i = 0; i < 8; ++i) {
                float p = __builtin_exp2f((c[i] + u) + t2[i]);  // normalized
                // streaming 536MB output: non-temporal, don't thrash L2
                __builtin_nontemporal_store(
                    p, aOut + (size_t)(i + 8 * hi) * NS + k0 + lo16);
                lp[wid][i + 8 * hi][sub * 16 + lo16] = (bf16_t)p;
            }
        }
        // P (row-major in LDS) read back in A-layout: wave-private region, DS
        // ops from one wave are in-order, compiler inserts the dscnt wait.
        const bf16_t* prow = &lp[wid][lo16][0];
        v16bf pa = cat(ld8(prow + hi * 16), ld8(prow + hi * 16 + 8));
        // V^T B-operands: lane n = dim column, keys contiguous in memory.
#pragma unroll
        for (int ch = 0; ch < 4; ++ch) {
            const bf16_t* vrow =
                vt + ((size_t)(bh * ND + ch * 16 + lo16)) * NS + k0p + hi * 16;
            v16bf vb = cat(ld8(vrow), ld8(vrow + 8));
            v8f& oc = (ch == 0) ? o0 : (ch == 1) ? o1 : (ch == 2) ? o2 : o3;
            oc = wmma_bf16(pa, vb, oc);
        }
    }

    // ---------------- store O (already normalized) ---------------------------
    float* oOut = outO + ((size_t)bh * NS + q0) * ND;
#pragma unroll
    for (int i = 0; i < 8; ++i) {
        const size_t r = (size_t)(i + 8 * hi) * ND + lo16;
        oOut[r +  0] = o0[i];
        oOut[r + 16] = o1[i];
        oOut[r + 32] = o2[i];
        oOut[r + 48] = o3[i];
    }
}

// ---------------------------------------------------------------------------
extern "C" void kernel_launch(void* const* d_in, const int* in_sizes, int n_in,
                              void* d_out, int out_size, void* d_ws, size_t ws_size,
                              hipStream_t stream) {
    const float* q    = (const float*)d_in[0];
    const float* k    = (const float*)d_in[1];
    const float* v    = (const float*)d_in[2];
    const float* sent = (const float*)d_in[3];
    const int*   mask = (const int*)d_in[4];

    const size_t N = (size_t)NB * NH * NS * ND;          // 4,194,304
    // workspace layout: qb | kb | vt (bf16, 8MB each) | wm (16KB) = ~24MB
    bf16_t* qb = (bf16_t*)d_ws;
    bf16_t* kb = qb + N;
    bf16_t* vt = kb + N;
    float*  wm = (float*)(vt + N);

    float* outO = (float*)d_out;                          // [B,H,S,D]
    float* outA = outO + N;                               // [B,H,S,S]

    prep_kernel<<<(unsigned)((N + 255) / 256), 256, 0, stream>>>(
        q, k, v, sent, mask, qb, kb, vt, wm);

    attn_fused<<<NB * NH * (NS / 64), 128, 0, stream>>>(
        qb, kb, vt, wm, outO, outA);
}
